// XbmBatchSoftmaxLoss_39779987096331
// MI455X (gfx1250) — matrix operand order repaired and checked by reading, not compiled, not run on hardware
//
#include <hip/hip_runtime.h>

// ---------------------------------------------------------------------------
// Problem constants (from the reference): N=4096 rows, D=768 features,
// scores = MARGIN - pairwise L2 distance, row log-softmax, diagonal NLL mean.
// ---------------------------------------------------------------------------
constexpr int N = 4096;
constexpr int D = 768;
constexpr float MARGIN = 50.0f;

constexpr int BT = 128;          // output tile edge (rows and cols per block)
constexpr int KT = 32;           // K step per WMMA (bf16 16x16x32)
constexpr int NK = D / KT;       // 24 K-steps
constexpr int NCB = N / BT;      // 32 column blocks -> 32 softmax partials/row
constexpr int LDSROW = 40;       // padded LDS row stride (elements) -> no bank conflicts

typedef __attribute__((ext_vector_type(16))) __bf16 v16bf;
typedef __attribute__((ext_vector_type(8)))  float  v8f;

union FragU { uint4 u[2]; v16bf f; };

__device__ __forceinline__ unsigned short f2bf(float x) {
    unsigned u = __builtin_bit_cast(unsigned, x);
    unsigned r = u + 0x7FFFu + ((u >> 16) & 1u);   // round-to-nearest-even
    return (unsigned short)(r >> 16);
}

// gfx1250 async copy: global -> LDS, 16 bytes per lane, tracked by ASYNCcnt.
// First operand is the per-lane LDS byte address (low 32 bits of the generic
// address of the __shared__ buffer: hardware adds LDS_BASE itself), second is
// the per-lane 64-bit global address, GV addressing mode.
__device__ __forceinline__ void async_ld16(unsigned lds_byte, const void* gaddr) {
    asm volatile("global_load_async_to_lds_b128 %0, %1, off"
                 :: "v"(lds_byte), "v"(gaddr)
                 : "memory");
}
// Async loads retire in order: <=4 outstanding proves the oldest 4 are done.
__device__ __forceinline__ void wait_async_le4() {
    asm volatile("s_wait_asynccnt 0x4" ::: "memory");
}
__device__ __forceinline__ void wait_async0() {
    asm volatile("s_wait_asynccnt 0x0" ::: "memory");
}

// ---------------------------------------------------------------------------
// Kernel 0: fp32 -> bf16 copies of A and P, plus exact fp32 row squared norms.
// One block per row; 192 threads active (768/4 float4 lanes).
// ---------------------------------------------------------------------------
__global__ __launch_bounds__(256) void prep_kernel(
    const float* __restrict__ A, const float* __restrict__ P,
    unsigned short* __restrict__ Abf, unsigned short* __restrict__ Pbf,
    float* __restrict__ a2, float* __restrict__ p2)
{
    __shared__ float sA[256];
    __shared__ float sP[256];
    const int i = blockIdx.x;
    const int t = threadIdx.x;
    float sa = 0.0f, sp = 0.0f;
    if (t < D / 4) {
        const int j = t * 4;
        float4 av = *(const float4*)(A + (size_t)i * D + j);
        float4 pv = *(const float4*)(P + (size_t)i * D + j);
        sa = av.x * av.x + av.y * av.y + av.z * av.z + av.w * av.w;
        sp = pv.x * pv.x + pv.y * pv.y + pv.z * pv.z + pv.w * pv.w;
        uint2 ua, up;
        ua.x = (unsigned)f2bf(av.x) | ((unsigned)f2bf(av.y) << 16);
        ua.y = (unsigned)f2bf(av.z) | ((unsigned)f2bf(av.w) << 16);
        up.x = (unsigned)f2bf(pv.x) | ((unsigned)f2bf(pv.y) << 16);
        up.y = (unsigned)f2bf(pv.z) | ((unsigned)f2bf(pv.w) << 16);
        *(uint2*)(Abf + (size_t)i * D + j) = ua;
        *(uint2*)(Pbf + (size_t)i * D + j) = up;
    }
    sA[t] = sa; sP[t] = sp;
    __syncthreads();
    for (int s = 128; s > 0; s >>= 1) {
        if (t < s) { sA[t] += sA[t + s]; sP[t] += sP[t + s]; }
        __syncthreads();
    }
    if (t == 0) { a2[i] = sA[0]; p2[i] = sP[0]; }
}

// ---------------------------------------------------------------------------
// Kernel 1: 128x128 tile of cross = A . P^T via v_wmma_f32_16x16x32_bf16,
// fused with per-row (max, sum-exp) softmax partials and diagonal capture.
// 8 waves / block: wave grid 4 (rows) x 2 (cols); each wave owns 32x64.
// Two-deep async-copy pipeline (ASYNCcnt): tiles kt and kt+1 in flight,
// s_wait_asynccnt 4 retires only the older tile, so tile kt+2's copy overlaps
// all of step kt+1's WMMAs. K-loop fully unrolled (24 steps).
// ---------------------------------------------------------------------------
__global__ __launch_bounds__(256) void tile_kernel(
    const unsigned short* __restrict__ Abf,
    const unsigned short* __restrict__ Pbf,
    const float* __restrict__ a2, const float* __restrict__ p2,
    float* __restrict__ pm, float* __restrict__ ps,
    float* __restrict__ diag)
{
    __shared__ __align__(16) unsigned short As[2 * BT * LDSROW];
    __shared__ __align__(16) unsigned short Ps[2 * BT * LDSROW];
    __shared__ float redM[BT][2];
    __shared__ float redS[BT][2];

    constexpr unsigned BUFB = BT * LDSROW * 2;   // bytes per LDS buffer (10240)

    const int bx = blockIdx.x, by = blockIdx.y;
    const int r0 = by * BT, c0 = bx * BT;
    const int t = threadIdx.x;
    const int w = t >> 5, lane = t & 31;
    const int wr = w >> 1, wc = w & 1;       // 4 row-groups x 2 col-groups
    const int lm = lane & 15, lh = lane >> 4;

    v8f acc[2][4];
    #pragma unroll
    for (int i = 0; i < 2; ++i)
        #pragma unroll
        for (int j = 0; j < 4; ++j) acc[i][j] = (v8f)0.0f;

    // tile-fill mapping: 512 16B chunks per tile, 2 per thread (A and P each)
    const int row0 = t >> 2,          cc0 = t & 3;
    const int row1 = (t + 256) >> 2,  cc1 = (t + 256) & 3;

    const unsigned short* gA0 = Abf + (size_t)(r0 + row0) * D + cc0 * 8;
    const unsigned short* gA1 = Abf + (size_t)(r0 + row1) * D + cc1 * 8;
    const unsigned short* gP0 = Pbf + (size_t)(c0 + row0) * D + cc0 * 8;
    const unsigned short* gP1 = Pbf + (size_t)(c0 + row1) * D + cc1 * 8;

    const unsigned lA0 = (unsigned)(size_t)&As[row0 * LDSROW + cc0 * 8];
    const unsigned lA1 = (unsigned)(size_t)&As[row1 * LDSROW + cc1 * 8];
    const unsigned lP0 = (unsigned)(size_t)&Ps[row0 * LDSROW + cc0 * 8];
    const unsigned lP1 = (unsigned)(size_t)&Ps[row1 * LDSROW + cc1 * 8];

    // prologue: async-copy tiles 0 and 1 (8 copies in flight)
    async_ld16(lA0, gA0);
    async_ld16(lA1, gA1);
    async_ld16(lP0, gP0);
    async_ld16(lP1, gP1);
    async_ld16(lA0 + BUFB, gA0 + KT);
    async_ld16(lA1 + BUFB, gA1 + KT);
    async_ld16(lP0 + BUFB, gP0 + KT);
    async_ld16(lP1 + BUFB, gP1 + KT);

    #pragma unroll
    for (int kt = 0; kt < NK; ++kt) {
        // Oldest 4 copies (tile kt) complete; tile kt+1 may still be in flight.
        if (kt + 1 < NK) wait_async_le4(); else wait_async0();
        __syncthreads();   // tile kt visible from every wave

        const unsigned short* Ac = &As[(kt & 1) * (BT * LDSROW)];
        const unsigned short* Pc = &Ps[(kt & 1) * (BT * LDSROW)];

        FragU afr[2], bfr[4];
        #pragma unroll
        for (int ti = 0; ti < 2; ++ti) {
            // A 16x32 bf16 layout: lane m, K chunks {lh*8 .. +7} and {lh*8+16 .. +7}
            const int ar = wr * 32 + ti * 16 + lm;
            afr[ti].u[0] = *(const uint4*)&Ac[ar * LDSROW + lh * 8];
            afr[ti].u[1] = *(const uint4*)&Ac[ar * LDSROW + lh * 8 + 16];
        }
        #pragma unroll
        for (int tj = 0; tj < 4; ++tj) {
            // B 32x16 bf16 layout: lane n = row of P, K chunks {lh*8 .. +7} and {lh*8+16 .. +7}
            const int br = wc * 64 + tj * 16 + lm;
            const uint4* bp = (const uint4*)&Pc[br * LDSROW + lh * 8];
            bfr[tj].u[0] = bp[0];
            bfr[tj].u[1] = bp[2];
        }
        #pragma unroll
        for (int ti = 0; ti < 2; ++ti)
            #pragma unroll
            for (int tj = 0; tj < 4; ++tj)
                acc[ti][tj] = __builtin_amdgcn_wmma_f32_16x16x32_bf16(
                    false, afr[ti].f, false, bfr[tj].f,
                    (short)0, acc[ti][tj], false, false);

        __syncthreads();   // all waves done reading buf[kt&1]
        if (kt + 2 < NK) { // refill it two steps ahead
            const unsigned nb = (unsigned)(kt & 1) * BUFB;
            const int ko = (kt + 2) * KT;
            async_ld16(lA0 + nb, gA0 + ko);
            async_ld16(lA1 + nb, gA1 + ko);
            async_ld16(lP0 + nb, gP0 + ko);
            async_ld16(lP1 + nb, gP1 + ko);
        }
    }

    // ---- epilogue: scores + per-row (max, sum-exp) over this wave's 64 cols
    float a2l[2][8], p2l[4];
    #pragma unroll
    for (int ti = 0; ti < 2; ++ti)
        #pragma unroll
        for (int v = 0; v < 8; ++v)
            a2l[ti][v] = a2[r0 + wr * 32 + ti * 16 + v + 8 * lh];
    #pragma unroll
    for (int tj = 0; tj < 4; ++tj)
        p2l[tj] = p2[c0 + wc * 64 + tj * 16 + lm];

    const bool diagBlock = (bx == by);
    float Mv[2][8], Sv[2][8];
    #pragma unroll
    for (int ti = 0; ti < 2; ++ti) {
        #pragma unroll
        for (int v = 0; v < 8; ++v) {
            const int rl = wr * 32 + ti * 16 + v + 8 * lh;   // local row (C layout)
            float s[4];
            #pragma unroll
            for (int tj = 0; tj < 4; ++tj) {
                float sq = a2l[ti][v] + p2l[tj] - 2.0f * acc[ti][tj][v];
                sq = fmaxf(sq, 0.0f);
                s[tj] = MARGIN - sqrtf(sq + 1e-12f);
            }
            if (diagBlock) {
                const int clb = wc * 64 + lm;                // local col for tj=0
                #pragma unroll
                for (int tj = 0; tj < 4; ++tj)
                    if (clb + tj * 16 == rl) diag[r0 + rl] = s[tj];
            }
            float m = fmaxf(fmaxf(s[0], s[1]), fmaxf(s[2], s[3]));
            #pragma unroll
            for (int mask = 1; mask <= 8; mask <<= 1)
                m = fmaxf(m, __shfl_xor(m, mask, 32));       // half-wave reduce
            float e = expf(s[0] - m) + expf(s[1] - m) + expf(s[2] - m) + expf(s[3] - m);
            #pragma unroll
            for (int mask = 1; mask <= 8; mask <<= 1)
                e += __shfl_xor(e, mask, 32);
            Mv[ti][v] = m; Sv[ti][v] = e;
        }
    }

    if (lm == 0) {   // lanes 0 and 16 publish their 16 rows each
        #pragma unroll
        for (int ti = 0; ti < 2; ++ti)
            #pragma unroll
            for (int v = 0; v < 8; ++v) {
                const int rl = wr * 32 + ti * 16 + v + 8 * lh;
                redM[rl][wc] = Mv[ti][v];
                redS[rl][wc] = Sv[ti][v];
            }
    }
    __syncthreads();
    if (t < BT) {
        const float m0 = redM[t][0], m1 = redM[t][1];
        const float M = fmaxf(m0, m1);
        const float S = redS[t][0] * expf(m0 - M) + redS[t][1] * expf(m1 - M);
        pm[(size_t)(r0 + t) * NCB + bx] = M;
        ps[(size_t)(r0 + t) * NCB + bx] = S;
    }
}

// ---------------------------------------------------------------------------
// Kernel 2: merge 32 (m,s) partials per row -> per-row loss = LSE - score_ii.
// One wave per row (NCB == 32 lanes).
// ---------------------------------------------------------------------------
__global__ __launch_bounds__(256) void rowlse_kernel(
    const float* __restrict__ pm, const float* __restrict__ ps,
    const float* __restrict__ diag, float* __restrict__ loss)
{
    const int row = blockIdx.x * 8 + (threadIdx.x >> 5);
    const int lane = threadIdx.x & 31;
    const float m = pm[(size_t)row * NCB + lane];
    const float s = ps[(size_t)row * NCB + lane];
    float M = m;
    #pragma unroll
    for (int mask = 1; mask <= 16; mask <<= 1)
        M = fmaxf(M, __shfl_xor(M, mask, 32));
    float e = s * expf(m - M);
    #pragma unroll
    for (int mask = 1; mask <= 16; mask <<= 1)
        e += __shfl_xor(e, mask, 32);
    if (lane == 0) loss[row] = (M + logf(e)) - diag[row];
}

// ---------------------------------------------------------------------------
// Kernel 3: mean of 4096 per-row losses -> scalar.
// ---------------------------------------------------------------------------
__global__ __launch_bounds__(256) void finalred_kernel(
    const float* __restrict__ loss, float* __restrict__ out)
{
    __shared__ float sh[256];
    const int t = threadIdx.x;
    float s = 0.0f;
    for (int k = t; k < N; k += 256) s += loss[k];
    sh[t] = s;
    __syncthreads();
    for (int st = 128; st > 0; st >>= 1) {
        if (t < st) sh[t] += sh[t + st];
        __syncthreads();
    }
    if (t == 0) out[0] = sh[0] * (1.0f / (float)N);
}

// ---------------------------------------------------------------------------
extern "C" void kernel_launch(void* const* d_in, const int* in_sizes, int n_in,
                              void* d_out, int out_size, void* d_ws, size_t ws_size,
                              hipStream_t stream) {
    (void)in_sizes; (void)n_in; (void)out_size; (void)ws_size;
    const float* A = (const float*)d_in[0];
    const float* P = (const float*)d_in[1];
    float* out = (float*)d_out;

    char* ws = (char*)d_ws;
    size_t off = 0;
    unsigned short* Abf = (unsigned short*)(ws + off); off += (size_t)N * D * 2;   // 6.29 MB
    unsigned short* Pbf = (unsigned short*)(ws + off); off += (size_t)N * D * 2;   // 6.29 MB
    float* a2   = (float*)(ws + off); off += (size_t)N * 4;
    float* p2   = (float*)(ws + off); off += (size_t)N * 4;
    float* pm   = (float*)(ws + off); off += (size_t)N * NCB * 4;
    float* ps   = (float*)(ws + off); off += (size_t)N * NCB * 4;
    float* diag = (float*)(ws + off); off += (size_t)N * 4;
    float* loss = (float*)(ws + off); off += (size_t)N * 4;                         // ~13.7 MB total

    prep_kernel<<<N, 256, 0, stream>>>(A, P, Abf, Pbf, a2, p2);
    tile_kernel<<<dim3(N / BT, N / BT), 256, 0, stream>>>(Abf, Pbf, a2, p2, pm, ps, diag);
    rowlse_kernel<<<N / 8, 256, 0, stream>>>(pm, ps, diag, loss);
    finalred_kernel<<<1, 256, 0, stream>>>(loss, out);
}
